// MultiOrderGraphLayer_54211077210420
// MI455X (gfx1250) — matrix operand order repaired and checked by reading, not compiled, not run on hardware
//
#include <hip/hip_runtime.h>
#include <hip/hip_fp16.h>
#include <stdint.h>

typedef __attribute__((ext_vector_type(16))) _Float16 v16h;
typedef __attribute__((ext_vector_type(8)))  _Float16 v8h;
typedef __attribute__((ext_vector_type(8)))  float    v8f;

#define N_NODES 50000
#define N_EDGES 600000
#define DIM     128

// Guaranteed-native f32 atomic add (no-return form, VFLAT encoding).
// Tracked by STOREcnt; kernel-end implicit wait-idle covers completion.
__device__ __forceinline__ void atomic_add_f32_native(float* p, float v) {
    asm volatile("global_atomic_add_f32 %0, %1, off" :: "v"(p), "v"(v) : "memory");
}

// ---------------- convert x: f32 -> f16 ----------------
__global__ void k_cvt_x(const float* __restrict__ x, _Float16* __restrict__ xh, int n) {
    int i = blockIdx.x * 256 + threadIdx.x;
    if (i < n) xh[i] = (_Float16)x[i];
}

// ---------------- pack W1/W2 into CDNA5 WMMA B-operand layout ----------------
// Bpack[w][ntile][kstep][lane*16 + j] = W[w][ kstep*32 + (j<8 ? (lane>=16?8:0) + j : 16 + (lane>=16?8:0) + j-8) ][ ntile*16 + (lane&15) ]
// (B 32x16 f16: lanes 0-15 hold K={0..7,16..23} of column N=lane, lanes 16-31 hold K={8..15,24..31} of N=lane-16)
__global__ void k_pack_w(const float* __restrict__ W1, const float* __restrict__ W2,
                         _Float16* __restrict__ Bpack) {
    int idx = blockIdx.x * 256 + threadIdx.x;            // 2*8*4*32*16 = 32768
    int j    =  idx        & 15;
    int lane = (idx >> 4)  & 31;
    int ks   = (idx >> 9)  & 3;
    int t    = (idx >> 11) & 7;
    int w    =  idx >> 14;
    int k = ks * 32 + ((lane >> 4) << 3) + (j < 8 ? j : 8 + j);
    int n = t * 16 + (lane & 15);
    const float* W = w ? W2 : W1;
    Bpack[idx] = (_Float16)W[k * DIM + n];
}

// ---------------- WMMA GEMM: H[wsel] = Xh * W[wsel], f16 in / f32 accum ----------------
// grid.x = N_NODES/16 row-blocks, grid.y = 2 (W1/W2); 256 threads = 8 waves, wave = N-tile.
__global__ void k_gemm_wmma(const _Float16* __restrict__ Xh,
                            const _Float16* __restrict__ Bpack,
                            float* __restrict__ H) {
    const int wave = threadIdx.x >> 5;
    const int lane = threadIdx.x & 31;
    const int hi   = lane >> 4;                 // 0: low half-lanes, 1: high
    const int mrow = blockIdx.x * 16 + (lane & 15);
    const int wsel = blockIdx.y;

    const _Float16* xrow  = Xh + (size_t)mrow * DIM;
    const _Float16* bbase = Bpack + ((size_t)wsel * 8 + wave) * (4 * 512) + lane * 16;

    v8f c = {};
#pragma unroll
    for (int ks = 0; ks < 4; ++ks) {
        // A 16x32 f16 layout: lanes 0-15: K = ks*32 + {0..7, 16..23}; lanes 16-31: +8 shift
        const int kb = ks * 32 + hi * 8;
        v8h a0 = *(const v8h*)(xrow + kb);
        v8h a1 = *(const v8h*)(xrow + kb + 16);
        v16h a = __builtin_shufflevector(a0, a1, 0,1,2,3,4,5,6,7,8,9,10,11,12,13,14,15);
        v16h b = *(const v16h*)(bbase + ks * 512);
        c = __builtin_amdgcn_wmma_f32_16x16x32_f16(false, a, false, b, (short)0, c, false, false);
    }

    // C/D f32 16x16: lane 0-15 -> N=lane, VGPR r -> M=r; lane 16-31 -> N=lane-16, M=r+8
    float* Hout = H + ((size_t)wsel * N_NODES + (size_t)blockIdx.x * 16) * DIM
                    + wave * 16 + (lane & 15);
    union { v8f v; float f[8]; } cu; cu.v = c;
#pragma unroll
    for (int r = 0; r < 8; ++r)
        Hout[(size_t)(r + 8 * hi) * DIM] = cu.f[r];
}

// ---------------- degree (self-loop => init 1), then dinv = rsqrt(deg) ----------------
__global__ void k_deg_init(int* __restrict__ deg) {
    int i = blockIdx.x * 256 + threadIdx.x;
    if (i < N_NODES) deg[i] = 1;
}
__global__ void k_deg_accum(const long long* __restrict__ dst, int* __restrict__ deg) {
    int e = blockIdx.x * 256 + threadIdx.x;
    if (e < N_EDGES) atomicAdd(&deg[(int)dst[e]], 1);
}
__global__ void k_dinv(const int* __restrict__ deg, float* __restrict__ dinv) {
    int i = blockIdx.x * 256 + threadIdx.x;
    if (i < N_NODES) dinv[i] = rsqrtf((float)deg[i]);
}

// ---------------- acc init with self-loop contribution: acc = H * dinv^2 ----------------
__global__ void k_acc_init(const float* __restrict__ H, const float* __restrict__ dinv,
                           float* __restrict__ acc) {
    size_t i = (size_t)blockIdx.x * 256 + threadIdx.x;   // covers 2*N*D
    int node = (int)((i >> 7) % N_NODES);
    float di = dinv[node];
    acc[i] = H[i] * di * di;
}

// ---------------- edge scatter: one wave per edge, 4 floats/lane, both layers ----------------
__global__ void k_scatter(const long long* __restrict__ src, const long long* __restrict__ dst,
                          const float* __restrict__ H, const float* __restrict__ dinv,
                          float* __restrict__ acc) {
    const int wave = threadIdx.x >> 5;
    const int lane = threadIdx.x & 31;
    const int e = blockIdx.x * 8 + wave;                 // 75000 * 8 == N_EDGES exactly
    const int s = (int)src[e];
    const int d = (int)dst[e];
    const float norm = dinv[s] * dinv[d];
    const int f = lane * 4;
    const size_t L2 = (size_t)N_NODES * DIM;

    float4 m1 = *(const float4*)(H + (size_t)s * DIM + f);
    float* a1 = acc + (size_t)d * DIM + f;
    atomic_add_f32_native(a1 + 0, m1.x * norm);
    atomic_add_f32_native(a1 + 1, m1.y * norm);
    atomic_add_f32_native(a1 + 2, m1.z * norm);
    atomic_add_f32_native(a1 + 3, m1.w * norm);

    float4 m2 = *(const float4*)(H + L2 + (size_t)s * DIM + f);
    float* a2 = acc + L2 + (size_t)d * DIM + f;
    atomic_add_f32_native(a2 + 0, m2.x * norm);
    atomic_add_f32_native(a2 + 1, m2.y * norm);
    atomic_add_f32_native(a2 + 2, m2.z * norm);
    atomic_add_f32_native(a2 + 3, m2.w * norm);
}

// ---------------- finalize: out = 0.5*(relu(acc1+b1) + relu(acc2+b2)) ----------------
__global__ void k_final(const float* __restrict__ acc, const float* __restrict__ b1,
                        const float* __restrict__ b2, float* __restrict__ out) {
    size_t i = (size_t)blockIdx.x * 256 + threadIdx.x;   // N*D
    const size_t L2 = (size_t)N_NODES * DIM;
    int f = (int)(i & (DIM - 1));
    float o1 = fmaxf(acc[i]      + b1[f], 0.0f);
    float o2 = fmaxf(acc[L2 + i] + b2[f], 0.0f);
    out[i] = 0.5f * (o1 + o2);
}

extern "C" void kernel_launch(void* const* d_in, const int* in_sizes, int n_in,
                              void* d_out, int out_size, void* d_ws, size_t ws_size,
                              hipStream_t stream) {
    const float*     x   = (const float*)d_in[0];
    const long long* ei  = (const long long*)d_in[1];   // [2, E] int64: row0 = src, row1 = dst
    const float*     W1  = (const float*)d_in[2];
    const float*     b1  = (const float*)d_in[3];
    const float*     W2  = (const float*)d_in[4];
    const float*     b2  = (const float*)d_in[5];
    float*           out = (float*)d_out;

    char* ws = (char*)d_ws;
    const size_t sz_xh  = (size_t)N_NODES * DIM * sizeof(_Float16);   // 12.8 MB
    const size_t sz_bp  = (size_t)2 * 8 * 4 * 512 * sizeof(_Float16); // 64 KB
    const size_t sz_H   = (size_t)2 * N_NODES * DIM * sizeof(float);  // 51.2 MB
    const size_t sz_acc = sz_H;                                       // 51.2 MB
    const size_t sz_deg = ((size_t)N_NODES * sizeof(int)   + 255) & ~(size_t)255;
    size_t o = 0;
    _Float16* xh    = (_Float16*)(ws + o); o += sz_xh;
    _Float16* bpack = (_Float16*)(ws + o); o += sz_bp;
    float*    H     = (float*)   (ws + o); o += sz_H;
    float*    acc   = (float*)   (ws + o); o += sz_acc;
    int*      deg   = (int*)     (ws + o); o += sz_deg;
    float*    dinv  = (float*)   (ws + o);

    const long long* src = ei;
    const long long* dst = ei + N_EDGES;

    k_cvt_x   <<<(N_NODES * DIM) / 256, 256, 0, stream>>>(x, xh, N_NODES * DIM);
    k_pack_w  <<<32768 / 256,            256, 0, stream>>>(W1, W2, bpack);
    k_gemm_wmma<<<dim3(N_NODES / 16, 2), 256, 0, stream>>>(xh, bpack, H);
    k_deg_init <<<(N_NODES + 255) / 256, 256, 0, stream>>>(deg);
    k_deg_accum<<<(N_EDGES + 255) / 256, 256, 0, stream>>>(dst, deg);
    k_dinv     <<<(N_NODES + 255) / 256, 256, 0, stream>>>(deg, dinv);
    k_acc_init <<<(2 * N_NODES * DIM) / 256, 256, 0, stream>>>(H, dinv, acc);
    k_scatter  <<<N_EDGES / 8,           256, 0, stream>>>(src, dst, H, dinv, acc);
    k_final    <<<(N_NODES * DIM) / 256, 256, 0, stream>>>(acc, b1, b2, out);
}
